// LstmDecoder_24833500906190
// MI455X (gfx1250) — matrix operand order repaired and checked by reading, not compiled, not run on hardware
//
#include <hip/hip_runtime.h>
#include <hip/hip_bf16.h>

// LSTM decoder, persistent-kernel design for MI455X (gfx1250, wave32).
//   B=64, T=2048, F=256, H=512.
// - 32 persistent WGs x 256 thr; WG w owns h columns [16w, 16w+16)
// - W_cat slice (64 rows x 768 bf16 = 96KB) staged ONCE into LDS via the
//   Tensor Data Mover; B-fragments come from ds_load_b128 for all 2048 steps
// - gates GEMM via v_wmma_f32_16x16x32_bf16 (fp32 accum); wave = row-tile x K-half,
//   4 gates per wave, 2-way K reduction folded into the LDS state update
// - first K-chunk peeled so accumulators start as the WMMA literal-0 C operand
// - h ping-pongs (contiguous halves of one buffer -> global_load, not flat)
// - software-pipelined A-fragment prefetch; phase-2 B fragments hoisted
// - one device-wide barrier per step (monotonic atomic counter)

#define B_  64
#define T_  2048
#define F_  256
#define H_  512
#define NWG 32

typedef __attribute__((ext_vector_type(16))) __bf16 v16bf;
typedef __attribute__((ext_vector_type(8)))  float  v8f;
typedef unsigned int v4u  __attribute__((ext_vector_type(4)));
typedef int          v8i_ __attribute__((ext_vector_type(8)));
typedef int          v4i_ __attribute__((ext_vector_type(4)));

union Frag {
    unsigned u[8];
    __bf16   h[16];
    v16bf    v;
};

// LDS layout (dynamic shared):
//   [0,        98304)  : Wcat slice, 64 rows x 768 bf16  (4 gate groups x 16 rows)
//   [98304,   131072)  : gate partials, 2 Khalves x 4 gates x 64 x 16 f32
//   [131072,  135168)  : cell state c, 64 x 16 f32
//   [135168,  143360)  : out-proj partials, 8 x 256 f32
#define LDSG_OFF   98304
#define LDSC_OFF   131072
#define LDSP_OFF   135168
#define SMEM_BYTES 143360

__device__ inline unsigned short bf16u(float f) {
    union { __bf16 h; unsigned short u; } r;
    r.h = (__bf16)f;
    return r.u;
}
__device__ inline unsigned packbf(float lo, float hi) {
    union { __bf16 h[2]; unsigned u; } r;
    r.h[0] = (__bf16)lo; r.h[1] = (__bf16)hi;
    return r.u;
}
__device__ inline float sigmoidf_(float x) { return 1.0f / (1.0f + __expf(-x)); }

// A fragment (16x32 bf16) from row-major bf16 memory, rows row0..row0+15,
// K=[k0,k0+32). lanes 0-15: M=lane, K k0..k0+7 (V0-3), k0+16..23 (V4-7);
// lanes 16-31: M=lane-16, K k0+8..15 and k0+24..31.
__device__ inline void loadA_bf16(Frag& A, const unsigned short* __restrict__ base,
                                  size_t ld, int row0, int k0, int lane) {
    const int m  = lane & 15;
    const int ko = (lane >> 4) << 3; // 0 or 8
    const unsigned short* p = base + (size_t)(row0 + m) * ld + (size_t)(k0 + ko);
    uint4 a0 = *(const uint4*)(p);      // -> V0..V3
    uint4 a1 = *(const uint4*)(p + 16); // -> V4..V7
    A.u[0] = a0.x; A.u[1] = a0.y; A.u[2] = a0.z; A.u[3] = a0.w;
    A.u[4] = a1.x; A.u[5] = a1.y; A.u[6] = a1.z; A.u[7] = a1.w;
}

// Same fragment from fp32 memory with on-the-fly RNE bf16 conversion.
__device__ inline void loadA_f32cvt(Frag& A, const float* __restrict__ base,
                                    size_t ld, int row0, int k0, int lane) {
    const int m  = lane & 15;
    const int ko = (lane >> 4) << 3;
    const float* p = base + (size_t)(row0 + m) * ld + (size_t)(k0 + ko);
    float4 a0 = *(const float4*)(p);
    float4 a1 = *(const float4*)(p + 4);
    float4 a2 = *(const float4*)(p + 16);
    float4 a3 = *(const float4*)(p + 20);
    A.h[0]=(__bf16)a0.x;  A.h[1]=(__bf16)a0.y;  A.h[2]=(__bf16)a0.z;  A.h[3]=(__bf16)a0.w;
    A.h[4]=(__bf16)a1.x;  A.h[5]=(__bf16)a1.y;  A.h[6]=(__bf16)a1.z;  A.h[7]=(__bf16)a1.w;
    A.h[8]=(__bf16)a2.x;  A.h[9]=(__bf16)a2.y;  A.h[10]=(__bf16)a2.z; A.h[11]=(__bf16)a2.w;
    A.h[12]=(__bf16)a3.x; A.h[13]=(__bf16)a3.y; A.h[14]=(__bf16)a3.z; A.h[15]=(__bf16)a3.w;
}

// B fragment (32x16 bf16, KxN): column n = row (col0+n) of row-major W[n][k].
__device__ inline void loadB_any(Frag& Bf, const unsigned short* __restrict__ base,
                                 size_t ld, int col0, int k0, int lane) {
    const int n  = lane & 15;
    const int ko = (lane >> 4) << 3; // 0 or 8
    const unsigned short* p = base + (size_t)(col0 + n) * ld + (size_t)(k0 + ko);
    uint4 b0 = *(const uint4*)(p);     // -> V0..V3
    uint4 b1 = *(const uint4*)(p + 16); // -> V4..V7
    Bf.u[0] = b0.x; Bf.u[1] = b0.y; Bf.u[2] = b0.z; Bf.u[3] = b0.w;
    Bf.u[4] = b1.x; Bf.u[5] = b1.y; Bf.u[6] = b1.z; Bf.u[7] = b1.w;
}

__device__ inline v8f wmma_bf16(const Frag& a, const Frag& b, v8f c) {
    return __builtin_amdgcn_wmma_f32_16x16x32_bf16(
        false, a.v, false, b.v, (short)0, c, false, false);
}

// Device-wide barrier: monotonic counter (no reset -> no reuse race).
// RELEASE RMW / ACQUIRE load provide the dev-scope wb/inv fencing.
__device__ inline void grid_barrier(unsigned* __restrict__ cnt, unsigned round) {
    __syncthreads();
    if (threadIdx.x == 0) {
        __hip_atomic_fetch_add(cnt, 1u, __ATOMIC_RELEASE, __HIP_MEMORY_SCOPE_AGENT);
        while (__hip_atomic_load(cnt, __ATOMIC_ACQUIRE, __HIP_MEMORY_SCOPE_AGENT)
               < round * NWG) {
            __builtin_amdgcn_s_sleep(1);
        }
    }
    __syncthreads();
}

#if __has_builtin(__builtin_amdgcn_tensor_load_to_lds)
// Stage `rows` contiguous rows of `row_dwords` DWORDs (stride == width) from
// global bf16 memory into LDS at byte offset lds_off, via the Tensor Data Mover.
__device__ inline void tdm_load_rows(unsigned lds_off, const unsigned short* gptr,
                                     int rows, int row_dwords) {
    const unsigned long long ga = (unsigned long long)(size_t)gptr;
    v4u g0;
    g0[0] = 1u;                                   // count=1, user descriptor
    g0[1] = lds_off;                              // D#.lds_addr
    g0[2] = (unsigned)(ga & 0xFFFFFFFFu);         // global_addr[31:0]
    g0[3] = (unsigned)((ga >> 32) & 0x01FFFFFFu)  // global_addr[56:32]
          | (2u << 30);                           // type=2 (image)
    v8i_ g1;
    g1[0] = (int)(2u << 16);                                       // data_size=4B
    g1[1] = (int)((unsigned)(row_dwords & 0xFFFF) << 16);          // tensor_dim0 lo
    g1[2] = (int)(((unsigned)row_dwords >> 16) |
                  ((unsigned)(rows & 0xFFFF) << 16));              // dim0 hi | dim1 lo
    g1[3] = (int)(((unsigned)rows >> 16) |
                  ((unsigned)(row_dwords & 0xFFFF) << 16));        // dim1 hi | tile_dim0
    g1[4] = (int)(rows & 0xFFFF);                                  // tile_dim1 (tile_dim2=0)
    g1[5] = (int)row_dwords;                                       // tensor_dim0_stride lo
    g1[6] = 0;
    g1[7] = 0;
    v4i_ z4 = {0, 0, 0, 0};
    v8i_ z8 = {0, 0, 0, 0, 0, 0, 0, 0};
    __builtin_amdgcn_tensor_load_to_lds(g0, g1, z4, z4, z8, 0);
}
#endif

// ---------------- prep: bf16 staging, biases, h0=z, barrier=0 ----------------
__global__ __launch_bounds__(256)
void lstm_prep(const float* __restrict__ z,
               const float* __restrict__ x,
               const float* __restrict__ W_ih,   // [4H, F]
               const float* __restrict__ W_hh,   // [4H, H]
               const float* __restrict__ b_ih,
               const float* __restrict__ b_hh,
               const float* __restrict__ W_lin,  // [F, H]
               unsigned short* __restrict__ Wcat, // [4H, F+H] bf16
               unsigned short* __restrict__ Wlin, // [F, H] bf16
               float* __restrict__ bias,          // [4H] = b_ih+b_hh
               unsigned short* __restrict__ hbase, // [2][B, H] bf16 (init half 0)
               unsigned short* __restrict__ xbf,  // [B, T, F] bf16 (may be null)
               unsigned* __restrict__ barcnt) {
    const size_t i0     = (size_t)blockIdx.x * blockDim.x + threadIdx.x;
    const size_t stride = (size_t)gridDim.x * blockDim.x;
    for (size_t i = i0; i < (size_t)(4 * H_) * (F_ + H_); i += stride) {
        const size_t n = i / (F_ + H_);
        const size_t c = i % (F_ + H_);
        const float v = (c < F_) ? W_ih[n * F_ + c] : W_hh[n * H_ + (c - F_)];
        Wcat[i] = bf16u(v);
    }
    for (size_t i = i0; i < (size_t)F_ * H_; i += stride)
        Wlin[i] = bf16u(W_lin[i]);
    for (size_t i = i0; i < (size_t)(4 * H_); i += stride)
        bias[i] = b_ih[i] + b_hh[i];
    for (size_t i = i0; i < (size_t)B_ * H_; i += stride)
        hbase[i] = bf16u(z[i]);
    if (xbf) {
        const float4* xf = (const float4*)x;
        uint4* xo = (uint4*)xbf;
        for (size_t i = i0; i < (size_t)B_ * T_ * F_ / 8; i += stride) {
            float4 a = xf[2 * i], b = xf[2 * i + 1];
            uint4 o;
            o.x = packbf(a.x, a.y); o.y = packbf(a.z, a.w);
            o.z = packbf(b.x, b.y); o.w = packbf(b.z, b.w);
            xo[i] = o;
        }
    }
    if (i0 == 0) barcnt[0] = 0u;
}

// ---------------- persistent recurrence kernel ----------------
__global__ __launch_bounds__(256, 1)
void lstm_persistent(const float* __restrict__ x,            // fp32 fallback
                     const unsigned short* __restrict__ xbf, // bf16 (may be null)
                     const float* __restrict__ bias,         // [4H]
                     const float* __restrict__ b_lin,        // [F]
                     const unsigned short* __restrict__ Wcat,
                     const unsigned short* __restrict__ Wlin,
                     unsigned short* __restrict__ hbase,     // [2][B,H] bf16, contiguous
                     unsigned* __restrict__ barcnt,
                     float* __restrict__ out) {
    extern __shared__ char smem[];
    unsigned short* lds_w = (unsigned short*)smem;           // [64][768] bf16
    float* lds_g = (float*)(smem + LDSG_OFF);                // [2][4][64][16]
    float* lds_c = (float*)(smem + LDSC_OFF);                // [64][16]
    float* lds_p = (float*)(smem + LDSP_OFF);                // [8][256]

    const int wg   = blockIdx.x;
    const int tid  = threadIdx.x;
    const int wave = tid >> 5;
    const int lane = tid & 31;
    const int hc0  = wg * 16;
    const int rt   = wave & 3;   // batch row tile
    const int kh   = wave >> 2;  // K half (0: k<384, 1: k>=384)
    const int mh   = (lane >> 4) * 8;
    const int nn   = lane & 15;

    // ---- one-time: stage this WG's Wcat slice (4 gate groups x 16 rows x 768)
#if __has_builtin(__builtin_amdgcn_tensor_load_to_lds)
    if (wave == 0) {
        const unsigned lds0 = (unsigned)(size_t)(void*)lds_w;
        #pragma unroll
        for (int g = 0; g < 4; ++g) {
            tdm_load_rows(lds0 + (unsigned)g * 16u * 768u * 2u,
                          Wcat + (size_t)(g * H_ + hc0) * (F_ + H_),
                          16, (F_ + H_) / 2 /* dwords per row */);
        }
#if __has_builtin(__builtin_amdgcn_s_wait_tensorcnt)
        __builtin_amdgcn_s_wait_tensorcnt(0);
#endif
    }
#else
    {
        unsigned* dst = (unsigned*)lds_w;
        for (int i = tid; i < 64 * 768 / 2; i += 256) {
            const int r = i / 384, c = i % 384;
            const int g = r >> 4, rr = r & 15;
            dst[i] = ((const unsigned*)(Wcat + (size_t)(g * H_ + hc0 + rr) * (F_ + H_)))[c];
        }
    }
#endif
    // zero persistent cell state
    #pragma unroll
    for (int j = 0; j < 4; ++j) {
        const int e = tid + 256 * j;
        lds_c[e] = 0.0f;
    }
    __syncthreads();

    // ---- phase-2 constants: hoist W_lin B fragments for the whole t loop
    const int tl   = wave >> 2;
    const int kq   = wave & 3;
    const int tile = wg * 2 + tl;
    const int ct   = tile & 15;
    const int rt2  = tile >> 4;
    Frag Bp0, Bp1, Bp2, Bp3;
    loadB_any(Bp0, Wlin, H_, ct * 16, kq * 128 + 0,  lane);
    loadB_any(Bp1, Wlin, H_, ct * 16, kq * 128 + 32, lane);
    loadB_any(Bp2, Wlin, H_, ct * 16, kq * 128 + 64, lane);
    loadB_any(Bp3, Wlin, H_, ct * 16, kq * 128 + 96, lane);

    unsigned round = 0;

    for (int t = 0; t < T_; ++t) {
        const unsigned short* hcur = hbase + (size_t)(t & 1) * (B_ * H_);
        unsigned short*       hnxt = hbase + (size_t)((t & 1) ^ 1) * (B_ * H_);

        // ---- phase 1: gates; this wave: 16 rows x all 4 gates x half of K
        const int kbeg = kh * 12, kend = kbeg + 12;
        const int xend = (kend < 8) ? kend : 8;

        // peel first K-chunk (kbeg): x-chunk 0 for kh=0, h-chunk 12 for kh=1,
        // so all four accumulators fold their init into the WMMA literal-0 C.
        v8f acc0, acc1, acc2, acc3;
        {
            Frag A0f;
            if (kh == 0) {
                if (xbf) loadA_bf16(A0f, xbf + (size_t)t * F_, (size_t)T_ * F_, rt * 16, 0, lane);
                else     loadA_f32cvt(A0f, x + (size_t)t * F_, (size_t)T_ * F_, rt * 16, 0, lane);
            } else {
                loadA_bf16(A0f, hcur, H_, rt * 16, (12 - 8) * 32, lane);
            }
            Frag B0, B1, B2, B3;
            loadB_any(B0, lds_w, F_ + H_, 0,  kbeg * 32, lane);
            loadB_any(B1, lds_w, F_ + H_, 16, kbeg * 32, lane);
            loadB_any(B2, lds_w, F_ + H_, 32, kbeg * 32, lane);
            loadB_any(B3, lds_w, F_ + H_, 48, kbeg * 32, lane);
            v8f zc = {};
            acc0 = wmma_bf16(A0f, B0, zc);
            acc1 = wmma_bf16(A0f, B1, zc);
            acc2 = wmma_bf16(A0f, B2, zc);
            acc3 = wmma_bf16(A0f, B3, zc);
        }

        // x chunks (K in feature range), pipelined A prefetch
        if (kbeg + 1 < xend) {
            Frag A_cur;
            if (xbf) loadA_bf16(A_cur, xbf + (size_t)t * F_, (size_t)T_ * F_, rt * 16, (kbeg + 1) * 32, lane);
            else     loadA_f32cvt(A_cur, x + (size_t)t * F_, (size_t)T_ * F_, rt * 16, (kbeg + 1) * 32, lane);
            for (int kc = kbeg + 1; kc < xend; ++kc) {
                const int kp = (kc + 1 < xend) ? (kc + 1) : kc;   // clamped prefetch
                Frag A_nxt;
                if (xbf) loadA_bf16(A_nxt, xbf + (size_t)t * F_, (size_t)T_ * F_, rt * 16, kp * 32, lane);
                else     loadA_f32cvt(A_nxt, x + (size_t)t * F_, (size_t)T_ * F_, rt * 16, kp * 32, lane);
                Frag B0, B1, B2, B3;
                loadB_any(B0, lds_w, F_ + H_, 0,  kc * 32, lane);
                loadB_any(B1, lds_w, F_ + H_, 16, kc * 32, lane);
                loadB_any(B2, lds_w, F_ + H_, 32, kc * 32, lane);
                loadB_any(B3, lds_w, F_ + H_, 48, kc * 32, lane);
                acc0 = wmma_bf16(A_cur, B0, acc0);
                acc1 = wmma_bf16(A_cur, B1, acc1);
                acc2 = wmma_bf16(A_cur, B2, acc2);
                acc3 = wmma_bf16(A_cur, B3, acc3);
                A_cur = A_nxt;
            }
        }

        // h chunks (K in hidden range), pipelined A prefetch
        const int hbeg = (kbeg + 1 > 8) ? (kbeg + 1) : 8;
        {
            Frag A_cur;
            loadA_bf16(A_cur, hcur, H_, rt * 16, (hbeg - 8) * 32, lane);
            #pragma unroll 2
            for (int kc = hbeg; kc < kend; ++kc) {
                const int kp = (kc + 1 < kend) ? (kc + 1) : kc;   // clamped prefetch
                Frag A_nxt;
                loadA_bf16(A_nxt, hcur, H_, rt * 16, (kp - 8) * 32, lane);
                Frag B0, B1, B2, B3;
                loadB_any(B0, lds_w, F_ + H_, 0,  kc * 32, lane);
                loadB_any(B1, lds_w, F_ + H_, 16, kc * 32, lane);
                loadB_any(B2, lds_w, F_ + H_, 32, kc * 32, lane);
                loadB_any(B3, lds_w, F_ + H_, 48, kc * 32, lane);
                acc0 = wmma_bf16(A_cur, B0, acc0);
                acc1 = wmma_bf16(A_cur, B1, acc1);
                acc2 = wmma_bf16(A_cur, B2, acc2);
                acc3 = wmma_bf16(A_cur, B3, acc3);
                A_cur = A_nxt;
            }
        }
        {
            float* gbase = lds_g + (size_t)kh * 4 * 64 * 16;
            const int rbase = (rt * 16 + mh) * 16 + nn;
            #pragma unroll
            for (int v = 0; v < 8; ++v) {
                gbase[0 * 64 * 16 + rbase + v * 16] = acc0[v];
                gbase[1 * 64 * 16 + rbase + v * 16] = acc1[v];
                gbase[2 * 64 * 16 + rbase + v * 16] = acc2[v];
                gbase[3 * 64 * 16 + rbase + v * 16] = acc3[v];
            }
        }
        __syncthreads();

        // ---- state update (sums the two K-half partials)
        #pragma unroll
        for (int j = 0; j < 4; ++j) {
            const int e = tid + 256 * j;   // 0..1023
            const int m = e >> 4, n = e & 15;
            const int col = hc0 + n;
            const int idx = m * 16 + n;
            const float gi = lds_g[(0 * 64 * 16) + idx] + lds_g[(4 + 0) * 64 * 16 + idx] + bias[col];
            const float gf = lds_g[(1 * 64 * 16) + idx] + lds_g[(4 + 1) * 64 * 16 + idx] + bias[H_ + col];
            const float gg = lds_g[(2 * 64 * 16) + idx] + lds_g[(4 + 2) * 64 * 16 + idx] + bias[2 * H_ + col];
            const float go = lds_g[(3 * 64 * 16) + idx] + lds_g[(4 + 3) * 64 * 16 + idx] + bias[3 * H_ + col];
            const float iv = sigmoidf_(gi);
            const float fv = sigmoidf_(gf);
            const float gv = tanhf(gg);
            const float ov = sigmoidf_(go);
            const float cn = fv * lds_c[idx] + iv * gv;
            lds_c[idx] = cn;
            hnxt[(size_t)m * H_ + col] = bf16u(ov * tanhf(cn));
        }

        ++round;
        grid_barrier(barcnt, round);

        // ---- phase 2: out_t = h_new @ Wlin^T (64 tiles, 4-way K split)
        // two accumulation chains -> loads/WMMAs can overlap; summed on store
        Frag A0, A1, A2, A3;
        loadA_bf16(A0, hnxt, H_, rt2 * 16, kq * 128 + 0,  lane);
        loadA_bf16(A1, hnxt, H_, rt2 * 16, kq * 128 + 32, lane);
        loadA_bf16(A2, hnxt, H_, rt2 * 16, kq * 128 + 64, lane);
        loadA_bf16(A3, hnxt, H_, rt2 * 16, kq * 128 + 96, lane);
        v8f zc2 = {};
        v8f pa = wmma_bf16(A0, Bp0, zc2);
        v8f pb = wmma_bf16(A1, Bp1, zc2);
        pa = wmma_bf16(A2, Bp2, pa);
        pb = wmma_bf16(A3, Bp3, pb);
        {
            float* lp = lds_p + (size_t)(tl * 4 + kq) * 256;
            #pragma unroll
            for (int v = 0; v < 8; ++v)
                lp[(mh + v) * 16 + nn] = pa[v] + pb[v];
        }
        __syncthreads();
        #pragma unroll
        for (int j = 0; j < 2; ++j) {
            const int tile2 = wg * 2 + j;
            const int ct2 = tile2 & 15, rt3 = tile2 >> 4;
            const float s = lds_p[(j * 4 + 0) * 256 + tid] + lds_p[(j * 4 + 1) * 256 + tid] +
                            lds_p[(j * 4 + 2) * 256 + tid] + lds_p[(j * 4 + 3) * 256 + tid];
            const int m = tid >> 4, n = tid & 15;
            const int b = rt3 * 16 + m;
            const int f = ct2 * 16 + n;
            out[(size_t)b * T_ * F_ + (size_t)t * F_ + f] = s + b_lin[f];
        }
        // lds_p rewritten only after the next grid_barrier's __syncthreads
    }
}

extern "C" void kernel_launch(void* const* d_in, const int* in_sizes, int n_in,
                              void* d_out, int out_size, void* d_ws, size_t ws_size,
                              hipStream_t stream) {
    (void)in_sizes; (void)n_in; (void)out_size;
    const float* z     = (const float*)d_in[0];
    const float* x     = (const float*)d_in[1];
    const float* W_ih  = (const float*)d_in[2];
    const float* W_hh  = (const float*)d_in[3];
    const float* b_ih  = (const float*)d_in[4];
    const float* b_hh  = (const float*)d_in[5];
    const float* W_lin = (const float*)d_in[6];
    const float* b_lin = (const float*)d_in[7];
    float* out = (float*)d_out;

    // Workspace layout (256B aligned):
    //   [0,256)        barrier counter
    //   [256,8448)     bias (b_ih+b_hh) f32
    //   [8448,..)      h ping+pong bf16 (2 x 64KB, contiguous)
    //   then           Wcat bf16 [2048x768] (3MiB), Wlin bf16 [256x512] (256KiB)
    //   then           x bf16 [64x2048x256] (64MiB) -- only if ws_size permits
    char* ws = (char*)d_ws;
    unsigned*       barcnt = (unsigned*)(ws + 0);
    float*          bias   = (float*)(ws + 256);
    unsigned short* hbase  = (unsigned short*)(ws + 8448);
    unsigned short* Wcat   = (unsigned short*)(ws + 139520);
    unsigned short* Wlin   = (unsigned short*)(ws + 3285248);
    const size_t FULL_NEED = 3547392u + 67108864u;
    unsigned short* xbf    = (ws_size >= FULL_NEED) ? (unsigned short*)(ws + 3547392)
                                                    : (unsigned short*)nullptr;

    lstm_prep<<<1024, 256, 0, stream>>>(z, x, W_ih, W_hh, b_ih, b_hh, W_lin,
                                        Wcat, Wlin, bias, hbase, xbf, barcnt);
    lstm_persistent<<<NWG, 256, SMEM_BYTES, stream>>>(x, xbf, bias, b_lin, Wcat, Wlin,
                                                      hbase, barcnt, out);
}
